// SparseStatAttention_83099027243234
// MI455X (gfx1250) — matrix operand order repaired and checked by reading, not compiled, not run on hardware
//
#include <hip/hip_runtime.h>
#include <hip/hip_bf16.h>

typedef __attribute__((ext_vector_type(16))) __bf16 v16bf;
typedef __attribute__((ext_vector_type(8)))  __bf16 v8bf;
typedef __attribute__((ext_vector_type(8)))  float  v8f;

union V16 { v16bf v; v8bf h[2]; };

static constexpr int  Bc = 2;
static constexpr int  Tc = 2048;
static constexpr int  Dc = 1024;
static constexpr int  Hc = 16;
static constexpr int  DHc = 64;
static constexpr int  VOCABc = 4096;
static constexpr float SCALEc = 0.125f;   // 1/sqrt(64)
static constexpr float ALPHAc = 0.5f;

__device__ __forceinline__ v8f wmma_bf16(v16bf a, v16bf b, v8f c) {
  return __builtin_amdgcn_wmma_f32_16x16x32_bf16(false, a, false, b, (short)0, c, false, false);
}

// ---------------------------------------------------------------- convert (flat)
__global__ void cvt_f32_bf16(const float* __restrict__ s, __bf16* __restrict__ d, int n) {
  int i = blockIdx.x * blockDim.x + threadIdx.x;
  if (i < n) d[i] = (__bf16)s[i];
}

// ------------------------------------------------- convert + transpose weights
// W[k][n] f32 (row-major KxN)  ->  Wt[n][k] bf16 (row-major NxK)
__global__ void cvt_transpose_bf16(const float* __restrict__ W, __bf16* __restrict__ Wt,
                                   int Kdim, int Ndim) {
  __shared__ float tile[32][33];
  const int nb = blockIdx.x * 32;
  const int kb = blockIdx.y * 32;
  const int tx = threadIdx.x & 31;
  const int ty = threadIdx.x >> 5;    // 0..7
  #pragma unroll
  for (int p = 0; p < 4; ++p) {
    int k = kb + ty + p * 8;
    tile[ty + p * 8][tx] = W[(size_t)k * Ndim + nb + tx];
  }
  __syncthreads();
  #pragma unroll
  for (int p = 0; p < 4; ++p) {
    int n = nb + ty + p * 8;
    Wt[(size_t)n * Kdim + kb + tx] = (__bf16)tile[tx][ty + p * 8];
  }
}

// ---------------------------------------------------------------- GEMM
// C[M,N] = A[M,K](bf16) @ W[K,N] + bias[N], with W given PRE-TRANSPOSED as
// Wt[N][K] bf16. Block: 256 thr = 8 waves (2x4), block tile 64x256,
// wave tile 32x64 (8 WMMA per k-chunk of 32).
__global__ void gemm_wmma_bf16(const __bf16* __restrict__ A,
                               const __bf16* __restrict__ Wt,
                               const float*  __restrict__ bias,
                               void* __restrict__ Cout,
                               int M, int N, int K, int out_bf16) {
  __shared__ __bf16 At[64 * 32];     // [row][k]   4 KB
  __shared__ __bf16 Bt[256 * 32];    // [col][k]  16 KB

  const int tid  = threadIdx.x;
  const int wave = tid >> 5;
  const int lane = tid & 31;
  const int lm   = lane & 15;
  const int hi   = lane >> 4;
  const int wm   = wave >> 2;        // 0..1 -> m offset wm*32
  const int wn   = wave & 3;         // 0..3 -> n offset wn*64
  const int m0   = blockIdx.y * 64;
  const int n0   = blockIdx.x * 256;

  v8f c[2][4] = {};

  for (int k0 = 0; k0 < K; k0 += 32) {
    { // stage A tile 64x32 (8 halfs/thread, contiguous b128 stores)
      int r = tid >> 2, cs = (tid & 3) * 8;
      *(v8bf*)&At[r * 32 + cs] = *(const v8bf*)&A[(size_t)(m0 + r) * K + k0 + cs];
    }
    { // stage B tile 256 cols x 32 k: one row of Wt per thread (64B)
      const __bf16* src = &Wt[(size_t)(n0 + tid) * K + k0];
      __bf16* dst = &Bt[tid * 32];
      #pragma unroll
      for (int p = 0; p < 4; ++p)
        *(v8bf*)(dst + p * 8) = *(const v8bf*)(src + p * 8);
    }
    __syncthreads();

    V16 a[2], bf[4];
    #pragma unroll
    for (int i = 0; i < 2; ++i) {
      const __bf16* ar = &At[(wm * 32 + i * 16 + lm) * 32];
      a[i].h[0] = *(const v8bf*)(ar + hi * 8);
      a[i].h[1] = *(const v8bf*)(ar + 16 + hi * 8);
    }
    #pragma unroll
    for (int j = 0; j < 4; ++j) {
      const __bf16* br = &Bt[(wn * 64 + j * 16 + lm) * 32];
      bf[j].h[0] = *(const v8bf*)(br + hi * 8);
      bf[j].h[1] = *(const v8bf*)(br + 16 + hi * 8);
    }
    #pragma unroll
    for (int i = 0; i < 2; ++i)
      #pragma unroll
      for (int j = 0; j < 4; ++j)
        c[i][j] = wmma_bf16(a[i].v, bf[j].v, c[i][j]);
    __syncthreads();
  }

  #pragma unroll
  for (int i = 0; i < 2; ++i) {
    #pragma unroll
    for (int j = 0; j < 4; ++j) {
      int col = n0 + wn * 64 + j * 16 + lm;
      float bv = bias[col];
      #pragma unroll
      for (int r = 0; r < 8; ++r) {
        int row = m0 + wm * 32 + i * 16 + r + hi * 8;
        float val = c[i][j][r] + bv;
        size_t off = (size_t)row * N + col;
        if (out_bf16) ((__bf16*)Cout)[off] = (__bf16)val;
        else          ((float*)Cout)[off]  = val;
      }
    }
  }
}

// ---------------------------------------------------------------- attention
// One block per (b, h, 16-query tile). 256 thr = 8 waves.
// Phase 1: WMMA scores (16 q x 2048 k) + pmi bias -> LDS bf16.
// Phase 2: per wave, 2 queries: iterative top-32 + softmax + attn@V.
__global__ void attn_topk_kernel(const __bf16* __restrict__ Q,
                                 const __bf16* __restrict__ Km,
                                 const float*  __restrict__ V,
                                 const int*    __restrict__ token_ids,
                                 const float*  __restrict__ pmi,
                                 __bf16* __restrict__ AO) {
  __shared__ __bf16 sc[16 * Tc];    // 64 KB

  const int qt   = blockIdx.x;
  const int h    = blockIdx.y;
  const int b    = blockIdx.z;
  const int q0   = qt * 16;
  const int tid  = threadIdx.x;
  const int wave = tid >> 5;
  const int lane = tid & 31;
  const int lm   = lane & 15;
  const int hi   = lane >> 4;

  // A fragments: Q tile 16x64 (k-chunks 0..31, 32..63)
  const __bf16* qrow = Q + ((size_t)(b * Tc + q0 + lm)) * Dc + h * DHc;
  V16 a0, a1;
  a0.h[0] = *(const v8bf*)(qrow + hi * 8);
  a0.h[1] = *(const v8bf*)(qrow + 16 + hi * 8);
  a1.h[0] = *(const v8bf*)(qrow + 32 + hi * 8);
  a1.h[1] = *(const v8bf*)(qrow + 48 + hi * 8);

  // Phase 1: each wave does 16 key tiles of 16 keys
  for (int j = 0; j < 16; ++j) {
    int k0 = (wave * 16 + j) * 16;
    const __bf16* krow = Km + ((size_t)(b * Tc + k0 + lm)) * Dc + h * DHc;
    V16 b0, b1;
    b0.h[0] = *(const v8bf*)(krow + hi * 8);
    b0.h[1] = *(const v8bf*)(krow + 16 + hi * 8);
    b1.h[0] = *(const v8bf*)(krow + 32 + hi * 8);
    b1.h[1] = *(const v8bf*)(krow + 48 + hi * 8);

    v8f c = {};
    c = wmma_bf16(a0.v, b0.v, c);
    c = wmma_bf16(a1.v, b1.v, c);

    int keycol = k0 + lm;
    int tokk = token_ids[b * Tc + keycol];
    #pragma unroll
    for (int r = 0; r < 8; ++r) {
      int m = r + hi * 8;
      int tokq = token_ids[b * Tc + q0 + m];
      float s = c[r] * SCALEc + ALPHAc * pmi[(size_t)tokq * VOCABc + tokk];
      sc[m * Tc + keycol] = (__bf16)s;
    }
  }
  __syncthreads();

  // Phase 2: wave owns queries 2*wave and 2*wave+1
  for (int qq = 0; qq < 2; ++qq) {
    const int m = wave * 2 + qq;
    __bf16* row = &sc[m * Tc];

    float myval = 0.0f; int myidx = 0;
    for (int iter = 0; iter < 32; ++iter) {
      float best = -1e30f; int bidx = 0x7fffffff;
      const v8bf* p8 = (const v8bf*)(row + lane * 64);
      for (int jj = 0; jj < 8; ++jj) {
        v8bf v8 = p8[jj];
        #pragma unroll
        for (int e = 0; e < 8; ++e) {
          float v = (float)v8[e];
          int idx = lane * 64 + jj * 8 + e;
          if (v > best) { best = v; bidx = idx; }
        }
      }
      #pragma unroll
      for (int off = 16; off > 0; off >>= 1) {
        float ov = __shfl_xor(best, off, 32);
        int   oi = __shfl_xor(bidx, off, 32);
        if (ov > best || (ov == best && oi < bidx)) { best = ov; bidx = oi; }
      }
      if (lane == iter) { myval = best; myidx = bidx; }
      if ((bidx >> 6) == lane) row[bidx] = (__bf16)(-1e30f);   // mask winner
    }

    // softmax over the 32 extracted values (lane i holds the i-th largest)
    float mx = __shfl(myval, 0, 32);
    float e  = __expf(myval - mx);
    float sum = e;
    #pragma unroll
    for (int off = 16; off > 0; off >>= 1) sum += __shfl_xor(sum, off, 32);
    float prob = e / sum;

    float* pf = (float*)row;        // reuse first 256B of this row
    int*   pi = (int*)row + 32;
    pf[lane] = prob;
    pi[lane] = myidx;

    // attn @ V : each lane handles dims lane and lane+32
    float acc0 = 0.0f, acc1 = 0.0f;
    for (int i = 0; i < 32; ++i) {
      float p  = pf[i];
      int  key = pi[i];
      const float* vrow = V + ((size_t)(b * Tc + key)) * Dc + h * DHc;
      acc0 += p * vrow[lane];
      acc1 += p * vrow[lane + 32];
    }
    __bf16* ao = AO + ((size_t)(b * Tc + q0 + m)) * Dc + h * DHc;
    ao[lane]      = (__bf16)acc0;
    ao[lane + 32] = (__bf16)acc1;
  }
}

// ---------------------------------------------------------------- launch
extern "C" void kernel_launch(void* const* d_in, const int* in_sizes, int n_in,
                              void* d_out, int out_size, void* d_ws, size_t ws_size,
                              hipStream_t stream) {
  const float* x         = (const float*)d_in[0];
  const int*   token_ids = (const int*)  d_in[1];
  const float* pmi       = (const float*)d_in[2];
  const float* wq = (const float*)d_in[3];
  const float* bq = (const float*)d_in[4];
  const float* wk = (const float*)d_in[5];
  const float* bk = (const float*)d_in[6];
  const float* wv = (const float*)d_in[7];
  const float* bv = (const float*)d_in[8];
  const float* wo = (const float*)d_in[9];
  const float* bo = (const float*)d_in[10];

  const int M = Bc * Tc;            // 4096
  const int N = Dc;                 // 1024
  const int K = Dc;                 // 1024

  char* w = (char*)d_ws;
  const size_t MB = 1024ull * 1024ull;
  __bf16* x16   = (__bf16*)(w);              //  8 MB
  __bf16* wq16t = (__bf16*)(w +  8 * MB);    //  2 MB each (transposed [N][K])
  __bf16* wk16t = (__bf16*)(w + 10 * MB);
  __bf16* wv16t = (__bf16*)(w + 12 * MB);
  __bf16* wo16t = (__bf16*)(w + 14 * MB);
  __bf16* Q16   = (__bf16*)(w + 16 * MB);    //  8 MB
  __bf16* K16   = (__bf16*)(w + 24 * MB);    //  8 MB
  float*  Vf    = (float*) (w + 32 * MB);    // 16 MB
  __bf16* AO16  = (__bf16*)(w + 48 * MB);    //  8 MB

  // convert x (flat) and weights (transpose+convert)
  {
    int n = M * Dc;
    cvt_f32_bf16<<<(n + 255) / 256, 256, 0, stream>>>(x, x16, n);
    dim3 tgrid(Dc / 32, Dc / 32);
    cvt_transpose_bf16<<<tgrid, 256, 0, stream>>>(wq, wq16t, K, N);
    cvt_transpose_bf16<<<tgrid, 256, 0, stream>>>(wk, wk16t, K, N);
    cvt_transpose_bf16<<<tgrid, 256, 0, stream>>>(wv, wv16t, K, N);
    cvt_transpose_bf16<<<tgrid, 256, 0, stream>>>(wo, wo16t, K, N);
  }

  dim3 ggrid(N / 256, M / 64);
  gemm_wmma_bf16<<<ggrid, 256, 0, stream>>>(x16, wq16t, bq, Q16, M, N, K, 1);
  gemm_wmma_bf16<<<ggrid, 256, 0, stream>>>(x16, wk16t, bk, K16, M, N, K, 1);
  gemm_wmma_bf16<<<ggrid, 256, 0, stream>>>(x16, wv16t, bv, Vf,  M, N, K, 0);

  dim3 agrid(Tc / 16, Hc, Bc);
  attn_topk_kernel<<<agrid, 256, 0, stream>>>(Q16, K16, Vf, token_ids, pmi, AO16);

  gemm_wmma_bf16<<<ggrid, 256, 0, stream>>>(AO16, wo16t, bo, (float*)d_out, M, N, K, 0);
}
